// VoronoiWatershedLoss_59614146068538
// MI455X (gfx1250) — compile-verified, matrix operand order checked by reading, not run on hardware
//
#include <hip/hip_runtime.h>
#include <hip/hip_bf16.h>

// ---------------------------------------------------------------------------
// Region erosion via chamfer distance transform, MI455X (gfx1250, wave32).
//   One workgroup (32 wave32 waves) per label: two raster passes with
//   closed-form horizontal min-plus relaxation (wave shuffle scans +
//   cross-wave aggregates). Marker rows are double-buffered into LDS with
//   CDNA5 async-to-LDS loads (ASYNCcnt). Label area is accumulated on the
//   matrix pipe with v_wmma_f32_16x16x32_f16. Threshold = exact k-th smallest
//   positive distance via 4-pass radix select on float bit patterns.
// ---------------------------------------------------------------------------

#define HH 1024
#define WW 1024
#define LL 24
#define HW (HH * WW)
#define CINF 1.0e9f
#define CA   1.0f
#define CB   1.41421356f

typedef __attribute__((ext_vector_type(16))) _Float16 v16h;
typedef __attribute__((ext_vector_type(8)))  float    v8f;

// ---- CDNA5 async global->LDS (ASYNCcnt) with compile-safe fallback --------
#if defined(__AMDGCN__) &&                                                    \
    __has_builtin(__builtin_amdgcn_global_load_async_to_lds_b32) &&           \
    __has_builtin(__builtin_amdgcn_s_wait_asynccnt)
#define HAVE_ASYNC_LDS 1
typedef __attribute__((address_space(1))) int gas_int;
typedef __attribute__((address_space(3))) int las_int;
#define ASYNC_ROW_LOAD(gptr, lptr)                                            \
    __builtin_amdgcn_global_load_async_to_lds_b32(                            \
        (gas_int*)(gptr), (las_int*)(lptr), 0, 0)
#define ASYNC_WAIT(nimm)                                                      \
    do { __builtin_amdgcn_s_wait_asynccnt(nimm);                              \
         asm volatile("" ::: "memory"); } while (0)
#else
#define HAVE_ASYNC_LDS 0
#define ASYNC_ROW_LOAD(gptr, lptr) do { *(lptr) = *(int*)(gptr); } while (0)
#define ASYNC_WAIT(nimm) do { asm volatile("" ::: "memory"); } while (0)
#endif

// One chamfer row relaxation: vertical relax from padded prev row (LDS) +
// closed-form horizontal min-plus closure. prevP has INF sentinels at [0] and
// [WW+1]; live entries at [1..WW] -> all three reads are unconditional.
__device__ __forceinline__ float row_relax(float d0, const float* prevP,
                                           float* aggA, float* aggB,
                                           int c, int lane, int wave) {
    float pv = prevP[c + 1] + CA;
    float pl = prevP[c + 2] + CB;
    float pr = prevP[c]     + CB;
    float v  = fminf(fminf(d0, pv), fminf(pl, pr));

    // left[k]  = k + min_{j<=k}(v_j - j)   (unit horizontal step weight)
    // right[k] = min_{j>=k}(v_j + j) - k
    float a = v - (float)c;
    float b = v + (float)c;
#pragma unroll
    for (int s = 1; s < 32; s <<= 1) {
        float ta = __shfl_up(a, (unsigned)s, 32);
        float tb = __shfl_down(b, (unsigned)s, 32);
        if (lane >= s)     a = fminf(a, ta);
        if (lane + s < 32) b = fminf(b, tb);
    }
    if (lane == 31) aggA[wave] = a;   // wave-total min of (v - j)
    if (lane == 0)  aggB[wave] = b;   // wave-total min of (v + j)
    __syncthreads();

    float accA = a, accB = b;
#pragma unroll 8
    for (int w2 = 0; w2 < 32; ++w2) {
        float va = aggA[w2];
        float vb = aggB[w2];
        accA = (w2 < wave) ? fminf(accA, va) : accA;
        accB = (w2 > wave) ? fminf(accB, vb) : accB;
    }
    return fminf(accA + (float)c, accB - (float)c);
}

// One workgroup (1024 threads) per label: forward + backward chamfer passes.
__global__ void __launch_bounds__(1024, 1)
dt_kernel(const int* __restrict__ markers, float* __restrict__ distBase,
          int* __restrict__ areaArr, int labelBase) {
    __shared__ float prevP[WW + 2];       // padded with INF sentinels
    __shared__ int   mrow[2][WW];         // async double-buffered marker rows
    __shared__ float aggA[32];
    __shared__ float aggB[32];
    __shared__ float areaAcc;

    const int c     = threadIdx.x;
    const int lane  = c & 31;
    const int wave  = c >> 5;
    const int label = labelBase + (int)blockIdx.x + 1;
    float* dist = distBase + (size_t)blockIdx.x * (size_t)HW;

    // ---- WMMA area accumulator: A = packed mask bits (f16), B = ones.
    v16h amask;
    v16h aones;
#pragma unroll
    for (int i = 0; i < 16; ++i) { aones[i] = (_Float16)1.0f; amask[i] = (_Float16)0.0f; }
    v8f cacc = {};

    prevP[c + 1] = CINF;
    if (c == 0) { prevP[0] = CINF; prevP[WW + 1] = CINF; }

    // Preload marker row 0 into LDS (lane i touches only its own slot, so
    // per-wave ASYNCcnt ordering is sufficient; no barrier needed for mrow).
    ASYNC_ROW_LOAD(markers + c, &mrow[0][c]);
    __syncthreads();

    // ---------------- forward pass (rows 0..H-1) ----------------
    for (int r = 0; r < HH; ++r) {
        const int cb = r & 1;
        if (r + 1 < HH) {
            ASYNC_ROW_LOAD(markers + (size_t)(r + 1) * WW + c, &mrow[cb ^ 1][c]);
            ASYNC_WAIT(1);               // row r resident, row r+1 in flight
        } else {
            ASYNC_WAIT(0);
        }
        bool  fg = (mrow[cb][c] == label);
        float d0 = fg ? CINF : 0.0f;

        amask[r & 15] = fg ? (_Float16)1.0f : (_Float16)0.0f;
        if ((r & 15) == 15) {
            // 512 mask elements per wave per issue; f32 accumulation is exact.
            cacc = __builtin_amdgcn_wmma_f32_16x16x32_f16(
                false, amask, false, aones, (short)0, cacc, false, false);
        }

        float nd = row_relax(d0, prevP, aggA, aggB, c, lane, wave);
        __syncthreads();                 // all prev/agg reads done
        prevP[c + 1] = nd;
        dist[(size_t)r * WW + c] = nd;
        __syncthreads();                 // prev visible before next row
    }

    // ---- reduce WMMA accumulator -> area. Sum over D (16x16) = 16 * sum(A).
    float tot = 0.0f;
#pragma unroll
    for (int i = 0; i < 8; ++i) tot += cacc[i];
#pragma unroll
    for (int s = 16; s > 0; s >>= 1) tot += __shfl_xor(tot, s, 32);
    if (threadIdx.x == 0) areaAcc = 0.0f;
    __syncthreads();
    if (lane == 0) atomicAdd(&areaAcc, tot);
    __syncthreads();
    if (threadIdx.x == 0)
        areaArr[labelBase + (int)blockIdx.x] = (int)(areaAcc * (1.0f / 16.0f) + 0.5f);

    // ---------------- backward pass (rows H-1..0) ----------------
    // dist was written with STOREcnt-tracked stores; re-read it with normal
    // loads (LOADcnt) -- async loads would be unordered vs those stores.
    prevP[c + 1] = CINF;
    __syncthreads();
    for (int r = HH - 1; r >= 0; --r) {
        if (r > 0) __builtin_prefetch(dist + (size_t)(r - 1) * WW + c, 0, 3);
        float d0 = dist[(size_t)r * WW + c];
        float nd = row_relax(d0, prevP, aggA, aggB, c, lane, wave);
        __syncthreads();
        prevP[c + 1] = nd;
        dist[(size_t)r * WW + c] = nd;
        __syncthreads();
    }
}

// Exact k-th-smallest positive distance via 4-pass radix select on float bits
// (positive IEEE floats order identically to their uint32 bit patterns).
__global__ void select_kernel(const float* __restrict__ distBase,
                              const int* __restrict__ areaArr,
                              const float* __restrict__ targets,
                              float* __restrict__ thrOut,
                              int* __restrict__ doErodeOut,
                              int labelBase) {
    __shared__ unsigned hist[256];
    __shared__ unsigned s_sel;
    __shared__ int s_k;

    const float* dist = distBase + (size_t)blockIdx.x * (size_t)HW;
    int   n      = areaArr[labelBase + (int)blockIdx.x];
    float target = targets[labelBase + (int)blockIdx.x];
    float nf     = (float)n;
    float areaf  = fmaxf(nf, 1.0f);
    float ratio  = 1.0f - target / areaf;
    int   thrIdx = (int)(nf * ratio);      // truncation matches astype(int32)
    bool  valid  = thrIdx < n;
    bool  erode  = (nf > target) && valid;

    if (!erode) {
        if (threadIdx.x == 0) { thrOut[blockIdx.x] = CINF; doErodeOut[blockIdx.x] = 0; }
        return;
    }

    int k = thrIdx; if (k < 0) k = 0; if (k > n - 1) k = n - 1;
    unsigned prefix = 0u;
    for (int byte = 3; byte >= 0; --byte) {
        if (threadIdx.x < 256) hist[threadIdx.x] = 0u;
        __syncthreads();
        unsigned hiMask = (byte == 3) ? 0u : (0xFFFFFFFFu << ((byte + 1) * 8));
        for (int i = (int)threadIdx.x; i < HW; i += (int)blockDim.x) {
            float d = dist[i];
            if (d > 0.0f) {
                unsigned u = __float_as_uint(d);
                if ((u & hiMask) == prefix)
                    atomicAdd(&hist[(u >> (byte * 8)) & 255u], 1u);
            }
        }
        __syncthreads();
        if (threadIdx.x == 0) {
            int cum = 0; unsigned sel = 255u; int kk = k;
            for (int b2 = 0; b2 < 256; ++b2) {
                int h = (int)hist[b2];
                if (kk < cum + h) { sel = (unsigned)b2; kk -= cum; break; }
                cum += h;
            }
            s_sel = sel; s_k = kk;
        }
        __syncthreads();
        prefix |= s_sel << (byte * 8);
        k = s_k;
        __syncthreads();
    }
    if (threadIdx.x == 0) {
        thrOut[blockIdx.x] = __uint_as_float(prefix);
        doErodeOut[blockIdx.x] = 1;
    }
}

// Zero out removed pixels; chunk 0 also initializes the float output.
__global__ void apply_kernel(const int* __restrict__ markers,
                             const float* __restrict__ distBase,
                             const float* __restrict__ thrArr,
                             const int* __restrict__ doErodeArr,
                             float* __restrict__ out,
                             int labelBase, int numLabels, int initOut) {
    int p = (int)(blockIdx.x * blockDim.x + threadIdx.x);
    if (p >= HW) return;
    int m = markers[p];
    float v = initOut ? (float)m : out[p];
    int s = m - 1 - labelBase;
    if (s >= 0 && s < numLabels && doErodeArr[s]) {
        float d = distBase[(size_t)s * (size_t)HW + p];
        if (d <= thrArr[s]) v = 0.0f;   // mask pixel with dist <= thr -> removed
    }
    out[p] = v;
}

extern "C" void kernel_launch(void* const* d_in, const int* in_sizes, int n_in,
                              void* d_out, int out_size, void* d_ws, size_t ws_size,
                              hipStream_t stream) {
    (void)in_sizes; (void)n_in; (void)out_size;
    const int*   markers = (const int*)d_in[0];
    const float* targets = (const float*)d_in[1];   // d_in[2] (image) unused by reference
    float* out = (float*)d_out;

    float* wsf       = (float*)d_ws;
    float* thrArr    = wsf;                 // [<=L] per-chunk slots
    int*   doErode   = (int*)(wsf + 32);    // [<=L]
    int*   areaArr   = (int*)(wsf + 64);    // [L], global index
    float* distBase  = wsf + 128;

    size_t wsFloats   = ws_size / sizeof(float);
    size_t availFloat = (wsFloats > 128) ? (wsFloats - 128) : 0;
    int maxC = (int)(availFloat / (size_t)HW);
    if (maxC < 1) maxC = 1;                 // assume >= ~4MB scratch
    if (maxC > LL) maxC = LL;               // 24 fields = 96MB -> resident in 192MB L2

    for (int base = 0; base < LL; base += maxC) {
        int C = (LL - base < maxC) ? (LL - base) : maxC;
        hipLaunchKernelGGL(dt_kernel, dim3(C), dim3(1024), 0, stream,
                           markers, distBase, areaArr, base);
        hipLaunchKernelGGL(select_kernel, dim3(C), dim3(256), 0, stream,
                           distBase, areaArr, targets, thrArr, doErode, base);
        hipLaunchKernelGGL(apply_kernel, dim3(HW / 256), dim3(256), 0, stream,
                           markers, distBase, thrArr, doErode, out,
                           base, C, (base == 0) ? 1 : 0);
    }
}
